// FinalCoarseToFineDensenSampleIGAModule_9182640078987
// MI455X (gfx1250) — matrix operand order repaired and checked by reading, not compiled, not run on hardware
//
#include <hip/hip_runtime.h>
#include <hip/hip_bf16.h>
#include <math.h>

// Problem constants (from reference)
#define BB 2
#define KK 512
#define CC 512
#define NN 4096
#define MM 24576            // OVERSAMPLE(6) * N
#define ALPHA  0.6f
#define SFLOOR 0.03f
#define SCEIL  2.0f
#define JITTER 1e-6f
#define BIGF   1e9f

typedef __attribute__((ext_vector_type(16))) __bf16 v16bf;
typedef __attribute__((ext_vector_type(8)))  __bf16 v8bf;
typedef __attribute__((ext_vector_type(8)))  float  v8f;

// ---------------- workspace layout (bytes) ----------------
#define OFF_CDF   0u                            // B*K f32      (4096 B)
#define OFF_LOGPI 4096u                         // B*K f32      (4096 B)
#define OFF_L     8192u                         // B*K*6 f32    (24576 B)
#define OFF_CAND  32768u                        // B*3*M f32    (589824 B)  SoA per batch
#define OFF_IDX   622592u                       // B*N i32      (32768 B)
#define OFF_WBF   655360u                       // B*N*K bf16   (8 MiB)
#define OFF_SPT   9043968u                      // B*C*K bf16   (1 MiB)   s_parent^T

// ---------------- output layout (floats) ----------------
#define OUT_S0   0u                             // B*N*C
#define OUT_MU0  (BB*NN*CC)                     // B*N*3
#define OUT_SIG0 (OUT_MU0 + BB*NN*3)            // B*N*9
#define OUT_W    (OUT_SIG0 + BB*NN*9)           // B*N*K

// =========================================================
// 1) pi, cdf, log(pi) per batch (one block of K threads)
// =========================================================
__global__ void pi_cdf_kernel(const float* __restrict__ occ,
                              const float* __restrict__ maskp,
                              float* __restrict__ cdf,
                              float* __restrict__ logpi) {
  __shared__ float arr[KK];
  const int b = blockIdx.x, tid = threadIdx.x;
  float p = occ[b * KK + tid] * (maskp[b * KK + tid] > 0.5f ? 1.0f : 0.0f);
  arr[tid] = p;
  __syncthreads();
  for (int off = 1; off < KK; off <<= 1) {           // inclusive scan
    float v = (tid >= off) ? arr[tid - off] : 0.0f;
    __syncthreads();
    arr[tid] += v;
    __syncthreads();
  }
  const float denom = fmaxf(arr[KK - 1], 1e-9f);
  cdf[b * KK + tid]   = arr[tid] / denom;
  logpi[b * KK + tid] = __logf(p / denom + 1e-9f);
}

// =========================================================
// 2) 3x3 Cholesky of Sig_p + 1e-6 I  (B*K threads)
// =========================================================
__global__ void chol_kernel(const float* __restrict__ Sig, float* __restrict__ Lm) {
  int i = blockIdx.x * blockDim.x + threadIdx.x;
  if (i >= BB * KK) return;
  const float* s = Sig + (size_t)i * 9;
  float l00 = sqrtf(s[0] + 1e-6f);
  float l10 = s[3] / l00;
  float l20 = s[6] / l00;
  float l11 = sqrtf(s[4] + 1e-6f - l10 * l10);
  float l21 = (s[7] - l20 * l10) / l11;
  float l22 = sqrtf(s[8] + 1e-6f - l20 * l20 - l21 * l21);
  float* L = Lm + (size_t)i * 6;
  L[0] = l00; L[1] = l10; L[2] = l11; L[3] = l20; L[4] = l21; L[5] = l22;
}

// =========================================================
// 3) candidates: comp = lower_bound(cdf, u); x = mu + L*eps
// =========================================================
__global__ void cand_kernel(const float* __restrict__ mu_p, const float* __restrict__ Lm,
                            const float* __restrict__ cdf,  const float* __restrict__ u_comp,
                            const float* __restrict__ eps_g, float* __restrict__ cand) {
  int i = blockIdx.x * blockDim.x + threadIdx.x;
  if (i >= BB * MM) return;
  const int b = i / MM, m = i % MM;
  const float u = u_comp[i];
  const float* cd = cdf + b * KK;
  int lo = 0, hi = KK;
  while (lo < hi) { int mid = (lo + hi) >> 1; if (cd[mid] < u) lo = mid + 1; else hi = mid; }
  const int comp = lo < (KK - 1) ? lo : (KK - 1);
  const size_t pk = (size_t)b * KK + comp;
  const float* mu = mu_p + pk * 3;
  const float* L  = Lm  + pk * 6;
  const float* e  = eps_g + (size_t)i * 3;
  float x = mu[0] + L[0] * e[0];
  float y = mu[1] + L[1] * e[0] + L[2] * e[1];
  float z = mu[2] + L[3] * e[0] + L[4] * e[1] + L[5] * e[2];
  float* cb = cand + (size_t)b * 3 * MM;
  cb[m] = x; cb[MM + m] = y; cb[2 * MM + m] = z;
}

// =========================================================
// 4) FPS: one workgroup per batch, whole point cloud in LDS
//    (3*M*4 = 288KB -- needs CDNA5's 320KB/WGP LDS)
// =========================================================
__global__ void __launch_bounds__(1024) fps_kernel(const float* __restrict__ cand,
                                                   int* __restrict__ idxg) {
  extern __shared__ float sm[];
  float* xs   = sm;
  float* ys   = sm + MM;
  float* zs   = sm + 2 * MM;
  float* redv = sm + 3 * MM;          // 32 floats
  int*   redi = (int*)(sm + 3 * MM + 32);
  float* bc   = sm + 3 * MM + 64;     // broadcast point
  const int b = blockIdx.x, tid = threadIdx.x;
  const float* cb = cand + (size_t)b * 3 * MM;
  for (int i = tid; i < MM; i += 1024) { xs[i] = cb[i]; ys[i] = cb[MM + i]; zs[i] = cb[2 * MM + i]; }
  __syncthreads();

  float d[MM / 1024];
  float px = xs[0], py = ys[0], pz = zs[0];
#pragma unroll
  for (int j = 0; j < MM / 1024; ++j) {
    int i = tid + j * 1024;
    float dx = xs[i] - px, dy = ys[i] - py, dz = zs[i] - pz;
    d[j] = dx * dx + dy * dy + dz * dz;
  }
  if (tid == 0) idxg[b * NN] = 0;

  for (int it = 1; it < NN; ++it) {
    float bv = -1.0f; int bi = 0;
#pragma unroll
    for (int j = 0; j < MM / 1024; ++j) {
      int i = tid + j * 1024;
      if (d[j] > bv) { bv = d[j]; bi = i; }
    }
    // wave32 argmax (lowest index wins ties, matching jnp.argmax)
#pragma unroll
    for (int off = 16; off; off >>= 1) {
      float ov = __shfl_xor(bv, off, 32);
      int   oi = __shfl_xor(bi, off, 32);
      if (ov > bv || (ov == bv && oi < bi)) { bv = ov; bi = oi; }
    }
    if ((tid & 31) == 0) { redv[tid >> 5] = bv; redi[tid >> 5] = bi; }
    __syncthreads();
    if (tid < 32) {
      bv = redv[tid]; bi = redi[tid];
#pragma unroll
      for (int off = 16; off; off >>= 1) {
        float ov = __shfl_xor(bv, off, 32);
        int   oi = __shfl_xor(bi, off, 32);
        if (ov > bv || (ov == bv && oi < bi)) { bv = ov; bi = oi; }
      }
      if (tid == 0) { bc[0] = xs[bi]; bc[1] = ys[bi]; bc[2] = zs[bi]; idxg[b * NN + it] = bi; }
    }
    __syncthreads();
    px = bc[0]; py = bc[1]; pz = bc[2];
#pragma unroll
    for (int j = 0; j < MM / 1024; ++j) {
      int i = tid + j * 1024;
      float dx = xs[i] - px, dy = ys[i] - py, dz = zs[i] - pz;
      d[j] = fminf(d[j], dx * dx + dy * dy + dz * dz);
    }
  }
}

// =========================================================
// 5) gather mu0
// =========================================================
__global__ void mu0_kernel(const float* __restrict__ cand, const int* __restrict__ idxg,
                           const float* __restrict__ node_mask, float* __restrict__ mu0) {
  int i = blockIdx.x * blockDim.x + threadIdx.x;
  if (i >= BB * NN) return;
  const int b = i / NN;
  const int sel = idxg[i];
  const float nm = node_mask[i];
  const float* cb = cand + (size_t)b * 3 * MM;
  mu0[(size_t)i * 3 + 0] = cb[sel] * nm;
  mu0[(size_t)i * 3 + 1] = cb[MM + sel] * nm;
  mu0[(size_t)i * 3 + 2] = cb[2 * MM + sel] * nm;
}

// =========================================================
// 6) 4-NN -> sigma -> Sig0 (LDS-tiled N x N)
// =========================================================
__global__ void knn_sig_kernel(const float* __restrict__ mu0,
                               const float* __restrict__ node_mask,
                               float* __restrict__ sig0) {
  __shared__ float tx[256], ty[256], tz[256], tm[256];
  const int b = blockIdx.x / (NN / 256);
  const int nb = blockIdx.x % (NN / 256);
  const int tid = threadIdx.x;
  const int n = nb * 256 + tid;
  const size_t node = (size_t)b * NN + n;
  const float x = mu0[node * 3], y = mu0[node * 3 + 1], z = mu0[node * 3 + 2];
  float b0 = BIGF, b1 = BIGF, b2 = BIGF, b3 = BIGF;   // 4 smallest, ascending
  for (int t0 = 0; t0 < NN; t0 += 256) {
    const size_t jn = (size_t)b * NN + t0 + tid;
    tx[tid] = mu0[jn * 3]; ty[tid] = mu0[jn * 3 + 1]; tz[tid] = mu0[jn * 3 + 2];
    tm[tid] = node_mask[jn];
    __syncthreads();
    for (int jj = 0; jj < 256; ++jj) {
      const int jg = t0 + jj;
      float dx = x - tx[jj], dy = y - ty[jj], dz = z - tz[jj];
      float d2 = dx * dx + dy * dy + dz * dz;
      if (jg == n || tm[jj] <= 0.5f) d2 = BIGF;
      if (d2 < b3) {
        if (d2 < b2) {
          b3 = b2;
          if (d2 < b1) { b2 = b1; if (d2 < b0) { b1 = b0; b0 = d2; } else b1 = d2; }
          else b2 = d2;
        } else b3 = d2;
      }
    }
    __syncthreads();
  }
  const float mean_d = 0.25f * (sqrtf(fmaxf(b0, 0.f)) + sqrtf(fmaxf(b1, 0.f)) +
                                sqrtf(fmaxf(b2, 0.f)) + sqrtf(fmaxf(b3, 0.f)));
  const float sigma = fminf(fmaxf(ALPHA * mean_d, SFLOOR), SCEIL);
  const float m = node_mask[node];
  const float dv = (sigma * sigma + JITTER) * m;
  float* o = sig0 + node * 9;
  o[0] = dv;  o[1] = 0.f; o[2] = 0.f;
  o[3] = 0.f; o[4] = dv;  o[5] = 0.f;
  o[6] = 0.f; o[7] = 0.f; o[8] = dv;
}

// =========================================================
// 7) softmax weights: one wave32 per node, 16 k's per lane
// =========================================================
__global__ void softmax_w_kernel(const float* __restrict__ mu0,
                                 const float* __restrict__ mu_p,
                                 const float* __restrict__ logpi,
                                 const float* __restrict__ mask_parent,
                                 const float* __restrict__ node_mask,
                                 float* __restrict__ w_out,
                                 __bf16* __restrict__ w_bf) {
  const int node = blockIdx.x * (blockDim.x >> 5) + (threadIdx.x >> 5);
  const int lane = threadIdx.x & 31;
  const int b = node / NN;
  const float x = mu0[(size_t)node * 3], y = mu0[(size_t)node * 3 + 1], z = mu0[(size_t)node * 3 + 2];
  float lg[16];
  float mx = -1e30f;
#pragma unroll
  for (int t = 0; t < 16; ++t) {
    const int k = lane + 32 * t;
    const size_t pk = (size_t)b * KK + k;
    float dx = x - mu_p[pk * 3], dy = y - mu_p[pk * 3 + 1], dz = z - mu_p[pk * 3 + 2];
    float v = logpi[pk] - 0.5f * (dx * dx + dy * dy + dz * dz);   // SIGMA_S = 1
    v = (mask_parent[pk] > 0.5f) ? v : -BIGF;
    lg[t] = v; mx = fmaxf(mx, v);
  }
#pragma unroll
  for (int off = 16; off; off >>= 1) mx = fmaxf(mx, __shfl_xor(mx, off, 32));
  float s = 0.f;
#pragma unroll
  for (int t = 0; t < 16; ++t) { float e = __expf(lg[t] - mx); lg[t] = e; s += e; }
#pragma unroll
  for (int off = 16; off; off >>= 1) s += __shfl_xor(s, off, 32);
  const float scale = node_mask[node] / s;
#pragma unroll
  for (int t = 0; t < 16; ++t) {
    const int k = lane + 32 * t;
    const float wv = lg[t] * scale;
    const size_t o = (size_t)node * KK + k;
    w_out[o] = wv;
    w_bf[o]  = (__bf16)wv;
  }
}

// =========================================================
// 8) s_parent -> bf16, transposed to [b][c][k]
// =========================================================
__global__ void spt_kernel(const float* __restrict__ sp, __bf16* __restrict__ spT) {
  int i = blockIdx.x * blockDim.x + threadIdx.x;
  if (i >= BB * KK * CC) return;
  const int b = i / (KK * CC);
  const int r = i % (KK * CC);
  const int k = r / CC, c = r % CC;
  spT[((size_t)b * CC + c) * KK + k] = (__bf16)sp[i];
}

// =========================================================
// 9) s0 = w @ s_parent via V_WMMA_F32_16X16X32_BF16
//    one wave per 16x16 tile; K-loop of 16 WMMAs
// =========================================================
__global__ void __launch_bounds__(256) gemm_wmma_kernel(const __bf16* __restrict__ wbf,
                                                        const __bf16* __restrict__ spT,
                                                        const float* __restrict__ node_mask,
                                                        float* __restrict__ s0) {
  const int wg = blockIdx.x * (blockDim.x >> 5) + (threadIdx.x >> 5);
  const int TN = NN / 16, TC = CC / 16;
  const int b  = wg / (TN * TC);
  if (b >= BB) return;                       // wave-uniform; EXEC stays all-ones
  const int r  = wg % (TN * TC);
  const int tn = r / TC, tc = r % TC;
  const int lane = threadIdx.x & 31;
  const int lm = lane & 15, half = lane >> 4;

  // A: w[b, tn*16+lm, :]   per ISA 16-bit A layout: K runs at half*8 and 16+half*8
  const __bf16* Ab = wbf + (size_t)b * NN * KK + (size_t)(tn * 16 + lm) * KK;
  // B: spT[b, tc*16+lm, :] same K order as A: K runs at half*8 and 16+half*8
  const __bf16* Bb = spT + (size_t)b * CC * KK + (size_t)(tc * 16 + lm) * KK;

  v8f acc = {};
  for (int kk = 0; kk < KK; kk += 32) {
    v8bf a0 = *(const v8bf*)(Ab + kk + half * 8);
    v8bf a1 = *(const v8bf*)(Ab + kk + 16 + half * 8);
    v8bf bq0 = *(const v8bf*)(Bb + kk + half * 8);
    v8bf bq1 = *(const v8bf*)(Bb + kk + 16 + half * 8);
    if (kk + 32 < KK) {                       // pull next K-slab toward the WGP
      __builtin_prefetch(Ab + kk + 32, 0, 1);
      __builtin_prefetch(Bb + kk + 32, 0, 1);
    }
    v16bf A, Bv;
#pragma unroll
    for (int e = 0; e < 8; ++e) { A[e] = a0[e]; A[e + 8] = a1[e]; Bv[e] = bq0[e]; Bv[e + 8] = bq1[e]; }
    acc = __builtin_amdgcn_wmma_f32_16x16x32_bf16(false, A, false, Bv, (short)0, acc, false, false);
  }
  const int c = tc * 16 + lm;
#pragma unroll
  for (int rr = 0; rr < 8; ++rr) {
    const int row = tn * 16 + rr + 8 * half;  // D layout: VGPR rr -> M = rr + 8*half
    const float nm = node_mask[b * NN + row];
    s0[((size_t)b * NN + row) * CC + c] = acc[rr] * nm;
  }
}

// =========================================================
// host launcher
// =========================================================
extern "C" void kernel_launch(void* const* d_in, const int* in_sizes, int n_in,
                              void* d_out, int out_size, void* d_ws, size_t ws_size,
                              hipStream_t stream) {
  (void)in_sizes; (void)n_in; (void)out_size; (void)ws_size;
  const float* s_parent    = (const float*)d_in[0];
  const float* mu_p        = (const float*)d_in[1];
  const float* Sig_p       = (const float*)d_in[2];
  const float* occ_parent  = (const float*)d_in[3];
  const float* mask_parent = (const float*)d_in[4];
  const float* node_mask   = (const float*)d_in[5];
  const float* u_comp      = (const float*)d_in[6];
  const float* eps_g       = (const float*)d_in[7];

  char* ws = (char*)d_ws;
  float*  ws_cdf   = (float*)(ws + OFF_CDF);
  float*  ws_logpi = (float*)(ws + OFF_LOGPI);
  float*  ws_L     = (float*)(ws + OFF_L);
  float*  ws_cand  = (float*)(ws + OFF_CAND);
  int*    ws_idx   = (int*)  (ws + OFF_IDX);
  __bf16* ws_wbf   = (__bf16*)(ws + OFF_WBF);
  __bf16* ws_spT   = (__bf16*)(ws + OFF_SPT);

  float* out   = (float*)d_out;
  float* o_s0  = out + OUT_S0;
  float* o_mu0 = out + OUT_MU0;
  float* o_sig = out + OUT_SIG0;
  float* o_w   = out + OUT_W;

  // 1) pi / cdf / log pi
  pi_cdf_kernel<<<BB, KK, 0, stream>>>(occ_parent, mask_parent, ws_cdf, ws_logpi);
  // 2) Cholesky
  chol_kernel<<<(BB * KK + 255) / 256, 256, 0, stream>>>(Sig_p, ws_L);
  // 3) candidates
  cand_kernel<<<(BB * MM + 255) / 256, 256, 0, stream>>>(mu_p, ws_L, ws_cdf, u_comp, eps_g, ws_cand);
  // 4) FPS: 288KB point cloud + reduction scratch in LDS (CDNA5: 320KB/WGP)
  const size_t fps_lds = (size_t)(3 * MM + 64 + 4) * sizeof(float);
  fps_kernel<<<BB, 1024, fps_lds, stream>>>(ws_cand, ws_idx);
  // 5) mu0 gather
  mu0_kernel<<<(BB * NN + 255) / 256, 256, 0, stream>>>(ws_cand, ws_idx, node_mask, o_mu0);
  // 6) KNN sigma -> Sig0
  knn_sig_kernel<<<BB * (NN / 256), 256, 0, stream>>>(o_mu0, node_mask, o_sig);
  // 7) softmax weights (f32 out + bf16 staging)
  softmax_w_kernel<<<(BB * NN) / 8, 256, 0, stream>>>(o_mu0, mu_p, ws_logpi, mask_parent,
                                                      node_mask, o_w, ws_wbf);
  // 8) s_parent -> bf16 transposed
  spt_kernel<<<(BB * KK * CC + 255) / 256, 256, 0, stream>>>(s_parent, ws_spT);
  // 9) WMMA GEMM: s0 = w @ s_parent
  const int tiles = BB * (NN / 16) * (CC / 16);     // 16384 waves
  gemm_wmma_kernel<<<tiles / 8, 256, 0, stream>>>(ws_wbf, ws_spT, node_mask, o_s0);
}
